// Net_32349693674203
// MI455X (gfx1250) — matrix operand order repaired and checked by reading, not compiled, not run on hardware
//
#include <hip/hip_runtime.h>
#include <math.h>

typedef __attribute__((ext_vector_type(16))) _Float16 v16h;
typedef __attribute__((ext_vector_type(8)))  _Float16 v8h;
typedef __attribute__((ext_vector_type(8)))  float    v8f;

#define B_   2
#define L_   2048
#define IN_  128
#define H_   512
#define OUT_ 64
#define NL_  2
#define ED_  1024
#define N_   16
#define DR_  32
#define K_   4
#define T_   (B_*L_)   /* 4096 tokens */
#define NBLK 4         /* N-tiles per wave (register blocking) */

__device__ __forceinline__ float sigmoidf_(float x) { return 1.0f / (1.0f + __expf(-x)); }

// ---------------------------------------------------------------------------
// WMMA GEMM: C[M,N] = act(A[M,K] * Bt^T + b1 + b2) + resid
//   - Bt: weights pre-transposed+converted to f16, Bt[N][K] -> B fragment is
//     two contiguous global_load_b128 per lane.
//   - A: f16 (two b128 loads) or f32 (converted on the fly; small-K GEMMs).
//   - Each wave computes a 16 x (16*NBLK) strip: one shared A fragment feeds
//     NBLK back-to-back v_wmma into NBLK accumulators (4x less A traffic).
// 8 waves/block; all tile counts divisible by 8*NBLK so EXEC is all ones
// around v_wmma (ISA requirement). ACT: 0=none,1=tanh,2=softplus,3=x-tanh(x)
// ---------------------------------------------------------------------------
template <int ACT, bool AF16, bool OF16, bool HB1, bool HB2, bool HRES>
__global__ __launch_bounds__(256)
void gemm_wmma(const void* __restrict__ Ap, int lda,
               const _Float16* __restrict__ Bt,
               void* __restrict__ Cp, int ldc,
               const float* __restrict__ bias1,
               const float* __restrict__ bias2,
               const float* __restrict__ resid,
               int tilesN, int Kdim)
{
  const int lane  = threadIdx.x & 31;
  const int wave  = threadIdx.x >> 5;
  const int group = blockIdx.x * 8 + wave;       // one 16x(16*NBLK) strip
  const int tilesN4 = tilesN / NBLK;
  const int tm  = group / tilesN4;
  const int tn0 = (group % tilesN4) * NBLK;
  const int hl = lane >> 4;     // lane half (0: lanes 0-15, 1: lanes 16-31)
  const int lr = lane & 15;
  const int rowA = tm * 16 + lr;

  const _Float16* __restrict__ Ah = (const _Float16*)Ap;
  const float*    __restrict__ Af = (const float*)Ap;
  const _Float16* __restrict__ Brow0 = Bt + (size_t)(tn0 * 16 + lr) * Kdim;

  v8f acc[NBLK];
  #pragma unroll
  for (int j = 0; j < NBLK; ++j) acc[j] = (v8f){};

  for (int kb = 0; kb < Kdim; kb += 32) {
    v16h af;
    if (AF16) {
      const v8h a0 = *(const v8h*)(Ah + (size_t)rowA * lda + kb + hl * 8);
      const v8h a1 = *(const v8h*)(Ah + (size_t)rowA * lda + kb + 16 + hl * 8);
      #pragma unroll
      for (int i = 0; i < 8; ++i) { af[i] = a0[i]; af[8 + i] = a1[i]; }
      // speculative prefetch of next K-chunk (silently dropped at tail)
      __builtin_prefetch(Ah + (size_t)rowA * lda + kb + 32 + hl * 8, 0, 0);
    } else {
      const float* ap = Af + (size_t)rowA * lda + kb + hl * 8;
      #pragma unroll
      for (int i = 0; i < 8; ++i) { af[i] = (_Float16)ap[i]; af[8 + i] = (_Float16)ap[16 + i]; }
      __builtin_prefetch(ap + 32, 0, 0);
    }
    // NBLK B fragments share the A fragment: bf[i] = Bt[colB_j][kb+hl*8+i], bf[8+i] = Bt[colB_j][kb+16+hl*8+i]
    #pragma unroll
    for (int j = 0; j < NBLK; ++j) {
      const _Float16* br = Brow0 + (size_t)(j * 16) * Kdim + kb + hl * 8;
      const v8h b0 = *(const v8h*)(br);
      const v8h b1 = *(const v8h*)(br + 16);
      v16h bf;
      #pragma unroll
      for (int i = 0; i < 8; ++i) { bf[i] = b0[i]; bf[8 + i] = b1[i]; }
      acc[j] = __builtin_amdgcn_wmma_f32_16x16x32_f16(false, af, false, bf,
                                                      (short)0, acc[j], false, false);
    }
  }

  #pragma unroll
  for (int j = 0; j < NBLK; ++j) {
    const int n = (tn0 + j) * 16 + lr;
    #pragma unroll
    for (int r = 0; r < 8; ++r) {
      const int m = tm * 16 + hl * 8 + r;   // C layout: VGPR r -> M = 8*half + r
      float v = acc[j][r];
      if (HB1) v += bias1[n];
      if (HB2) v += bias2[n];
      if (ACT == 1)      v = tanhf(v);
      else if (ACT == 2) v = (v > 20.0f) ? v : __logf(1.0f + __expf(v));
      else if (ACT == 3) v = v - tanhf(v);
      if (HRES) v += resid[(size_t)m * ldc + n];
      if (OF16) ((_Float16*)Cp)[(size_t)m * ldc + n] = (_Float16)v;
      else      ((float*)Cp)[(size_t)m * ldc + n] = v;
    }
  }
}

// ---------------------------------------------------------------------------
// Weight convert + transpose: W[K][N] f32 -> Wt[N][K] f16.
// ---------------------------------------------------------------------------
__global__ __launch_bounds__(256)
void cvt_transpose_kernel(const float* __restrict__ W, _Float16* __restrict__ Wt,
                          int Nn, int Kd)
{
  const int idx = blockIdx.x * 256 + threadIdx.x;   // over Kd*Nn (multiple of 256)
  const int k = idx / Nn;
  const int n = idx % Nn;
  Wt[(size_t)n * Kd + k] = (_Float16)W[idx];
}

// ---------------------------------------------------------------------------
// RMSNorm over H=512, one block (256 threads) per token; f16 output (GEMM A).
// ---------------------------------------------------------------------------
__global__ __launch_bounds__(256)
void rmsnorm_kernel(const float* __restrict__ x, const float* __restrict__ w,
                    _Float16* __restrict__ y)
{
  __shared__ float red[8];
  const float* xr = x + (size_t)blockIdx.x * H_;
  float ss = 0.f;
  for (int j = threadIdx.x; j < H_; j += 256) { float v = xr[j]; ss += v * v; }
  #pragma unroll
  for (int off = 16; off > 0; off >>= 1) ss += __shfl_down(ss, off, 32);
  if ((threadIdx.x & 31) == 0) red[threadIdx.x >> 5] = ss;
  __syncthreads();
  if (threadIdx.x < 8) {
    float v = red[threadIdx.x];
    #pragma unroll
    for (int off = 4; off > 0; off >>= 1) v += __shfl_down(v, off, 8);
    if (threadIdx.x == 0) red[0] = v;
  }
  __syncthreads();
  const float inv = rsqrtf(red[0] / (float)H_ + 1e-5f);
  _Float16* yr = y + (size_t)blockIdx.x * H_;
  for (int j = threadIdx.x; j < H_; j += 256) yr[j] = (_Float16)(xr[j] * inv * w[j]);
}

// ---------------------------------------------------------------------------
// Depthwise causal conv (K=4) + SiLU. Reads first ED columns of xz (stride
// 2*ED, f32), writes dense f16 xc[T,ED] (consumed by xproj GEMM and scan).
// ---------------------------------------------------------------------------
__global__ __launch_bounds__(256)
void convsilu_kernel(const float* __restrict__ xz, const float* __restrict__ w,
                     const float* __restrict__ bconv, _Float16* __restrict__ xc)
{
  const int idx = blockIdx.x * 256 + threadIdx.x;     // over T_*ED_
  const int d  = idx % ED_;
  const int bt = idx / ED_;
  const int t  = bt % L_;
  const int b  = bt / L_;
  const float* wr = w + d * K_;
  float acc = bconv[d];
  #pragma unroll
  for (int k = 0; k < K_; ++k) {
    const int tp = t - (K_ - 1) + k;
    if (tp >= 0) acc += xz[((size_t)(b * L_ + tp)) * (2 * ED_) + d] * wr[k];
  }
  xc[(size_t)bt * ED_ + d] = (_Float16)(acc * sigmoidf_(acc));
}

// ---------------------------------------------------------------------------
// Sequential SSM scan. One thread per (b,d) chain, N=16 state in registers.
// Fuses D*xc and the silu(z) gate; emits f16 gated y (GEMM A of out-proj).
// ---------------------------------------------------------------------------
__global__ __launch_bounds__(256)
void ssm_scan_kernel(const _Float16* __restrict__ xc, const float* __restrict__ delta,
                     const float* __restrict__ dbl, const float* __restrict__ xz,
                     const float* __restrict__ A_log, const float* __restrict__ Dp,
                     _Float16* __restrict__ yg)
{
  const int idx = blockIdx.x * 256 + threadIdx.x;     // over B_*ED_
  const int d = idx % ED_;
  const int b = idx / ED_;
  float a[N_], h[N_];
  #pragma unroll
  for (int n = 0; n < N_; ++n) { a[n] = -__expf(A_log[d * N_ + n]); h[n] = 0.f; }
  const float Dv = Dp[d];
  for (int t = 0; t < L_; ++t) {
    const size_t bt = (size_t)b * L_ + t;
    const float dv = delta[bt * ED_ + d];
    const float xv = (float)xc[bt * ED_ + d];
    const float zv = xz[bt * (2 * ED_) + ED_ + d];
    const float* dblr = dbl + bt * (DR_ + 2 * N_);
    const float dx = dv * xv;
    float y = 0.f;
    #pragma unroll
    for (int n = 0; n < N_; ++n) {
      const float dA = __expf(dv * a[n]);
      h[n] = dA * h[n] + dx * dblr[DR_ + n];          // B_t
      y += h[n] * dblr[DR_ + N_ + n];                 // C_t
    }
    y += Dv * xv;
    y *= zv * sigmoidf_(zv);
    yg[bt * ED_ + d] = (_Float16)y;
  }
}

// ---------------------------------------------------------------------------
template <int ACT, bool AF16, bool OF16, bool HB1, bool HB2, bool HRES>
static void launch_gemm(const void* A, int lda, const _Float16* Bt,
                        void* C, int ldc, const float* b1, const float* b2,
                        const float* res, int M, int Nn, int Kdim, hipStream_t s)
{
  const int tilesN = Nn / 16;
  const int tiles  = (M / 16) * tilesN;
  gemm_wmma<ACT, AF16, OF16, HB1, HB2, HRES>
      <<<dim3(tiles / (8 * NBLK)), dim3(256), 0, s>>>(A, lda, Bt, C, ldc,
                                                      b1, b2, res, tilesN, Kdim);
}

static void cvtw(const float* W, _Float16* Wt, int Kd, int Nn, hipStream_t s)
{
  cvt_transpose_kernel<<<dim3((Kd * Nn) / 256), dim3(256), 0, s>>>(W, Wt, Nn, Kd);
}

extern "C" void kernel_launch(void* const* d_in, const int* in_sizes, int n_in,
                              void* d_out, int out_size, void* d_ws, size_t ws_size,
                              hipStream_t stream)
{
  (void)in_sizes; (void)n_in; (void)out_size; (void)ws_size;
  const float* x        = (const float*)d_in[0];
  const float* p_Wi     = (const float*)d_in[1];
  const float* p_bi     = (const float*)d_in[2];
  const float* p_bh     = (const float*)d_in[3];
  const float* p_Wo     = (const float*)d_in[4];
  const float* p_bo     = (const float*)d_in[5];
  const float* m_norm_w = (const float*)d_in[6];
  const float* m_in_W   = (const float*)d_in[7];
  const float* m_conv_w = (const float*)d_in[8];
  const float* m_conv_b = (const float*)d_in[9];
  const float* m_xproj_W= (const float*)d_in[10];
  const float* m_dt_W   = (const float*)d_in[11];
  const float* m_dt_b   = (const float*)d_in[12];
  const float* m_A_log  = (const float*)d_in[13];
  const float* m_D      = (const float*)d_in[14];
  const float* m_out_W  = (const float*)d_in[15];
  const float* d_Wi     = (const float*)d_in[16];
  const float* d_bi     = (const float*)d_in[17];
  const float* d_bh     = (const float*)d_in[18];
  const float* d_Wo     = (const float*)d_in[19];
  const float* d_bo     = (const float*)d_in[20];
  float* out_f = (float*)d_out;

  // ---- workspace layout: f32 region then f16 region (~89 MB total) ----
  float* ws    = (float*)d_ws;
  float* outb  = ws;                                  // T*H
  float* xz    = outb  + (size_t)T_ * H_;             // T*2*ED
  float* dblb  = xz    + (size_t)T_ * 2 * ED_;        // T*64
  float* delta = dblb  + (size_t)T_ * (DR_ + 2 * N_); // T*ED

  _Float16* xn_h    = (_Float16*)(delta + (size_t)T_ * ED_);
  _Float16* xc_h    = xn_h   + (size_t)T_ * H_;       // T*ED
  _Float16* yg_h    = xc_h   + (size_t)T_ * ED_;      // T*ED
  _Float16* hid_h   = yg_h   + (size_t)T_ * ED_;      // T*H   (liquid1 hidden)
  _Float16* hid64_h = hid_h  + (size_t)T_ * H_;       // T*64  (liquid2 hidden)
  _Float16* wt      = hid64_h + (size_t)T_ * OUT_;

  _Float16* WiT    = wt;                               // [H][IN]
  _Float16* WoT    = WiT  + (size_t)IN_ * H_;          // [H][H]
  _Float16* inWT   = WoT  + (size_t)H_ * H_;           // NL x [2ED][H]
  _Float16* xprojT = inWT + (size_t)NL_ * H_ * 2 * ED_;// NL x [64][ED]
  _Float16* dtWT   = xprojT + (size_t)NL_ * ED_ * (DR_ + 2 * N_); // NL x [ED][DR]
  _Float16* outWT  = dtWT + (size_t)NL_ * DR_ * ED_;   // NL x [H][ED]
  _Float16* dWiT   = outWT + (size_t)NL_ * ED_ * H_;   // [64][H]
  _Float16* dWoT   = dWiT + (size_t)H_ * OUT_;         // [64][64]

  // ---- 0) convert + transpose all weights to f16 (L2-resident afterwards) ----
  cvtw(p_Wi, WiT, IN_, H_, stream);
  cvtw(p_Wo, WoT, H_, H_, stream);
  for (int l = 0; l < NL_; ++l) {
    cvtw(m_in_W    + (size_t)l * H_ * 2 * ED_,          inWT   + (size_t)l * H_ * 2 * ED_,          H_,  2 * ED_,        stream);
    cvtw(m_xproj_W + (size_t)l * ED_ * (DR_ + 2 * N_),  xprojT + (size_t)l * ED_ * (DR_ + 2 * N_),  ED_, DR_ + 2 * N_,   stream);
    cvtw(m_dt_W    + (size_t)l * DR_ * ED_,             dtWT   + (size_t)l * DR_ * ED_,             DR_, ED_,            stream);
    cvtw(m_out_W   + (size_t)l * ED_ * H_,              outWT  + (size_t)l * ED_ * H_,              ED_, H_,             stream);
  }
  cvtw(d_Wi, dWiT, H_, OUT_, stream);
  cvtw(d_Wo, dWoT, OUT_, OUT_, stream);

  // ---- 1) liquid input: hid = tanh(x@Wi + bi + bh) [f16]; out = hid@Wo + bo ----
  launch_gemm<1, false, true, true, true, false>(x, IN_, WiT, hid_h, H_,
                                                 p_bi, p_bh, nullptr, T_, H_, IN_, stream);
  launch_gemm<0, true, false, true, false, false>(hid_h, H_, WoT, outb, H_,
                                                  p_bo, nullptr, nullptr, T_, H_, H_, stream);

  // ---- 2) Mamba blocks ----
  for (int l = 0; l < NL_; ++l) {
    const float* conv_w = m_conv_w + (size_t)l * ED_ * K_;
    const float* conv_b = m_conv_b + (size_t)l * ED_;
    const float* dt_b   = m_dt_b   + (size_t)l * ED_;
    const float* A_log  = m_A_log  + (size_t)l * ED_ * N_;
    const float* Dp     = m_D      + (size_t)l * ED_;
    const float* norm_w = m_norm_w + (size_t)l * H_;

    rmsnorm_kernel<<<dim3(T_), dim3(256), 0, stream>>>(outb, norm_w, xn_h);
    // xz = xn @ in_W  (4096x512 @ 512x2048)
    launch_gemm<0, true, false, false, false, false>(xn_h, H_,
        inWT + (size_t)l * H_ * 2 * ED_, xz, 2 * ED_,
        nullptr, nullptr, nullptr, T_, 2 * ED_, H_, stream);
    // xc = silu(causal_conv(xz[:, :ED]))  [f16]
    convsilu_kernel<<<dim3((T_ * ED_) / 256), dim3(256), 0, stream>>>(xz, conv_w, conv_b, xc_h);
    // dbl = xc @ xproj_W  (4096x1024 @ 1024x64)
    launch_gemm<0, true, false, false, false, false>(xc_h, ED_,
        xprojT + (size_t)l * ED_ * (DR_ + 2 * N_), dblb, DR_ + 2 * N_,
        nullptr, nullptr, nullptr, T_, DR_ + 2 * N_, ED_, stream);
    // delta = softplus(dt @ dt_W + dt_b); dt = dbl[:, :32] via lda=64 (f32 A)
    launch_gemm<2, false, false, true, false, false>(dblb, DR_ + 2 * N_,
        dtWT + (size_t)l * DR_ * ED_, delta, ED_,
        dt_b, nullptr, nullptr, T_, ED_, DR_, stream);
    // sequential scan + gate -> f16 y_gated
    ssm_scan_kernel<<<dim3((B_ * ED_) / 256), dim3(256), 0, stream>>>(
        xc_h, delta, dblb, xz, A_log, Dp, yg_h);
    // out = out + y_gated @ out_W  (4096x1024 @ 1024x512, residual fused)
    launch_gemm<0, true, false, false, false, true>(yg_h, ED_,
        outWT + (size_t)l * ED_ * H_, outb, H_,
        nullptr, nullptr, outb, T_, H_, ED_, stream);
  }

  // ---- 3) liquid output + (x - tanh(x)) ----
  launch_gemm<1, false, true, true, true, false>(outb, H_, dWiT, hid64_h, OUT_,
                                                 d_bi, d_bh, nullptr, T_, OUT_, H_, stream);
  launch_gemm<3, true, false, true, false, false>(hid64_h, OUT_, dWoT, out_f, OUT_,
                                                  d_bo, nullptr, nullptr, T_, OUT_, OUT_, stream);
}
